// WindowAttention_59751585022208
// MI455X (gfx1250) — matrix operand order repaired and checked by reading, not compiled, not run on hardware
//
#include <hip/hip_runtime.h>
#include <hip/hip_bf16.h>
#include <stdint.h>

// ---------------- constants ----------------
#define NUM_HEADS 32
#define HEAD_DIM  32
#define NTOK      49   // window tokens
#define DIM       1024
#define B_        1024
#define NW        64
#define MTOT      (B_ * NTOK)   // 50176

typedef __attribute__((ext_vector_type(16))) __bf16 v16bf;
typedef __attribute__((ext_vector_type(8)))  float  v8f;

union FragBF {
    v16bf v;
    uint4 q[2];
};

union Pack8BF {
    __bf16 h[8];
    uint4  q;
};

__device__ __forceinline__ v8f wmma_bf16(const FragBF& a, const FragBF& b, v8f c) {
    return __builtin_amdgcn_wmma_f32_16x16x32_bf16(
        /*neg_a=*/false, a.v, /*neg_b=*/false, b.v,
        /*c_mod=*/(short)0, c, /*reuse_a=*/false, /*reuse_b=*/false);
}

// ---- CDNA5 async global->LDS copy (ASYNCcnt-tracked, per-lane 32B) ----
__device__ __forceinline__ void async_copy_32B(uint32_t lds_addr, const void* gptr) {
    asm volatile(
        "global_load_async_to_lds_b128 %0, %1, off\n\t"
        "global_load_async_to_lds_b128 %0, %1, off offset:16"
        :
        : "v"(lds_addr), "v"(gptr)
        : "memory");
}
__device__ __forceinline__ void wait_async_le4() {
    asm volatile("s_wait_asynccnt 0x4" ::: "memory");
}
__device__ __forceinline__ void wait_async_0() {
    asm volatile("s_wait_asynccnt 0x0" ::: "memory");
}

// ---------------------------------------------------------------
// fp32 -> bf16 elementwise (8 elems / thread, 16B stores)
// ---------------------------------------------------------------
__global__ void k_f32_to_bf16(const float* __restrict__ src,
                              __bf16* __restrict__ dst, int n8)
{
    int i = blockIdx.x * blockDim.x + threadIdx.x;
    if (i >= n8) return;
    const float4* s = (const float4*)src + (size_t)2 * i;
    float4 a = s[0], b = s[1];
    Pack8BF p;
    p.h[0] = (__bf16)a.x; p.h[1] = (__bf16)a.y; p.h[2] = (__bf16)a.z; p.h[3] = (__bf16)a.w;
    p.h[4] = (__bf16)b.x; p.h[5] = (__bf16)b.y; p.h[6] = (__bf16)b.z; p.h[7] = (__bf16)b.w;
    ((uint4*)dst)[i] = p.q;
}

// out_w (c,o) fp32 -> dst (o,c) bf16  (transpose + convert; 4 MB, one-shot)
__global__ void k_transpose_bf16(const float* __restrict__ src,
                                 __bf16* __restrict__ dst)
{
    int idx = blockIdx.x * blockDim.x + threadIdx.x;
    if (idx >= DIM * DIM) return;
    int o = idx / DIM, c = idx % DIM;
    dst[idx] = (__bf16)src[(size_t)c * DIM + o];
}

// gather relative position bias -> biasg[h][i][j] fp32
__global__ void k_bias_gather(const float* __restrict__ table,   // (169, 32)
                              const int*   __restrict__ rpidx,   // (49, 49)
                              float*       __restrict__ biasg)   // (32, 49, 49)
{
    int idx = blockIdx.x * blockDim.x + threadIdx.x;
    const int tot = NUM_HEADS * NTOK * NTOK;
    if (idx >= tot) return;
    int h  = idx / (NTOK * NTOK);
    int ij = idx % (NTOK * NTOK);
    biasg[idx] = table[rpidx[ij] * NUM_HEADS + h];
}

// ---------------------------------------------------------------
// Tiled WMMA GEMM  C[m][o] = sum_k A[m][k]*B[o][k] + bias[o]
//   A: bf16 (M x K) row-major, B: bf16 (O x K) row-major
//   BM=BN=128, BK=32, 256 threads (8 waves: 4 M-groups x 2 N-groups)
//   Double-buffered LDS filled by async global->LDS copies.
//   grid = (O/128, M/128), x fastest => weights stay L2-resident.
// ---------------------------------------------------------------
template<bool STORE_BF16>
__global__ __launch_bounds__(256) void k_gemm_wmma(
    const __bf16* __restrict__ A,
    const __bf16* __restrict__ Bw,
    const float*  __restrict__ bias,
    void*         __restrict__ Cptr,
    int M, int K, int O)
{
    __shared__ __align__(16) __bf16 As[2][128][32];
    __shared__ __align__(16) __bf16 Bs[2][128][32];

    const int oBase = blockIdx.x * 128;
    const int mBase = blockIdx.y * 128;
    const int tid   = threadIdx.x;
    const int lane  = tid & 31;
    const int wave  = tid >> 5;
    const int wm    = wave >> 1;     // 0..3 : 32-row group
    const int wn    = wave & 1;      // 0..1 : 64-col group
    const int l15   = lane & 15;
    const int hi    = lane >> 4;

    v8f acc[2][4];
    for (int t = 0; t < 2; ++t)
        for (int u = 0; u < 4; ++u)
            for (int i = 0; i < 8; ++i) acc[t][u][i] = 0.0f;

    const int lrow = tid >> 1;          // 0..127
    const int lch  = (tid & 1) * 16;    // 0 or 16

    const __bf16* gA = A  + (size_t)(mBase + lrow) * K + lch;
    const __bf16* gB = Bw + (size_t)(oBase + lrow) * K + lch;

    const int nK = K >> 5;

    // prologue: stage K-step 0 into buffer 0
    async_copy_32B((uint32_t)(uintptr_t)&As[0][lrow][lch], gA);
    async_copy_32B((uint32_t)(uintptr_t)&Bs[0][lrow][lch], gB);

    for (int i = 0; i < nK; ++i) {
        const int cur = i & 1;
        if (i + 1 < nK) {
            const int kk = (i + 1) << 5;
            async_copy_32B((uint32_t)(uintptr_t)&As[1 - cur][lrow][lch], gA + kk);
            async_copy_32B((uint32_t)(uintptr_t)&Bs[1 - cur][lrow][lch], gB + kk);
            wait_async_le4();     // step i resident (async loads complete in order)
        } else {
            wait_async_0();
        }
        __syncthreads();          // all waves' copies for step i visible

        FragBF af[2], bf[4];
        #pragma unroll
        for (int t = 0; t < 2; ++t) {
            const int row = wm * 32 + t * 16 + l15;                // A: lane = row M
            af[t].q[0] = *(const uint4*)&As[cur][row][hi * 8];      // K = hi*8..+7
            af[t].q[1] = *(const uint4*)&As[cur][row][16 + hi * 8]; // K = 16+hi*8..+7
        }
        #pragma unroll
        for (int u = 0; u < 4; ++u) {
            const int col = wn * 64 + u * 16 + l15;                // B: lane = col N
            bf[u].q[0] = *(const uint4*)&Bs[cur][col][hi * 8];      // K = hi*8..+7
            bf[u].q[1] = *(const uint4*)&Bs[cur][col][16 + hi * 8]; // K = 16+hi*8..+7
        }
        #pragma unroll
        for (int t = 0; t < 2; ++t)
            #pragma unroll
            for (int u = 0; u < 4; ++u)
                acc[t][u] = wmma_bf16(af[t], bf[u], acc[t][u]);
        __syncthreads();          // protect buffer reuse before next issue
    }

    // epilogue: + bias, store
    #pragma unroll
    for (int t = 0; t < 2; ++t) {
        #pragma unroll
        for (int u = 0; u < 4; ++u) {
            const int oc = oBase + wn * 64 + u * 16 + l15;
            const float bv = bias[oc];
            #pragma unroll
            for (int r = 0; r < 8; ++r) {
                const int mr = mBase + wm * 32 + t * 16 + hi * 8 + r;
                const float val = acc[t][u][r] + bv;
                if (STORE_BF16)
                    ((__bf16*)Cptr)[(size_t)mr * O + oc] = (__bf16)val;
                else
                    ((float*)Cptr)[(size_t)mr * O + oc] = val;
            }
        }
    }
}

// ---------------------------------------------------------------
// Fused window attention, one wave per (b, h)
//   qkv: (50176, 3072) bf16  [q | k | v] per row
//   o  : (50176, 1024) bf16
// ---------------------------------------------------------------
__global__ __launch_bounds__(128) void k_attention(
    const __bf16* __restrict__ qkv,
    const float*  __restrict__ mask,   // (64, 49, 49)
    const float*  __restrict__ biasg,  // (32, 49, 49)
    __bf16*       __restrict__ o)
{
    __shared__ __align__(16) __bf16 sP [4][64][64];  // softmax probs, row-major
    __shared__ __align__(16) __bf16 sVT[4][32][64];  // V transposed: [d][key]

    const int wave = threadIdx.x >> 5;
    const int lane = threadIdx.x & 31;
    const int b    = blockIdx.x >> 3;
    const int h    = ((blockIdx.x & 7) << 2) | wave;
    const int w    = b & (NW - 1);
    const int l15  = lane & 15;
    const int hi   = lane >> 4;
    const float scl = 0.17677669529663687f;  // HEAD_DIM^-0.5

    // ---- stage V transposed into LDS (zero-pad keys 49..63) ----
    for (int key = lane; key < 64; key += 32) {
        if (key < NTOK) {
            const __bf16* vrow = qkv + ((size_t)(b * NTOK + key)) * 3072 + 2048 + h * 32;
            #pragma unroll
            for (int d = 0; d < 32; ++d) sVT[wave][d][key] = vrow[d];
        } else {
            #pragma unroll
            for (int d = 0; d < 32; ++d) sVT[wave][d][key] = (__bf16)0.0f;
        }
    }

    // ---- S = Q * K^T  (4x4 tiles of 16x16, K=32 in one WMMA) ----
    v8f S[4][4];
    for (int t = 0; t < 4; ++t)
        for (int u = 0; u < 4; ++u)
            for (int i = 0; i < 8; ++i) S[t][u][i] = 0.0f;

    FragBF qf[4];
    #pragma unroll
    for (int t = 0; t < 4; ++t) {
        int r = t * 16 + l15; if (r > NTOK - 1) r = NTOK - 1;   // clamp pad rows
        const __bf16* qrow = qkv + ((size_t)(b * NTOK + r)) * 3072 + h * 32;
        qf[t].q[0] = *(const uint4*)(qrow + hi * 8);
        qf[t].q[1] = *(const uint4*)(qrow + 16 + hi * 8);
    }
    #pragma unroll
    for (int u = 0; u < 4; ++u) {
        int n = u * 16 + l15; if (n > NTOK - 1) n = NTOK - 1;
        const __bf16* krow = qkv + ((size_t)(b * NTOK + n)) * 3072 + 1024 + h * 32;
        FragBF kf;
        kf.q[0] = *(const uint4*)(krow + hi * 8);
        kf.q[1] = *(const uint4*)(krow + 16 + hi * 8);
        #pragma unroll
        for (int t = 0; t < 4; ++t)
            S[t][u] = wmma_bf16(qf[t], kf, S[t][u]);
    }

    // ---- softmax over rows (scale + bias + mask, pad cols -> -inf) ----
    #pragma unroll
    for (int t = 0; t < 4; ++t) {
        #pragma unroll
        for (int r = 0; r < 8; ++r) {
            const int i = t * 16 + hi * 8 + r;        // query row
            const bool irow = (i < NTOK);
            float lg[4];
            #pragma unroll
            for (int u = 0; u < 4; ++u) {
                const int j = u * 16 + l15;            // key col
                float x = S[t][u][r] * scl;
                if (irow && j < NTOK)
                    x += biasg[(h * NTOK + i) * NTOK + j]
                       + mask[((size_t)w * NTOK + i) * NTOK + j];
                else
                    x = -1e30f;
                lg[u] = x;
            }
            float mx = fmaxf(fmaxf(lg[0], lg[1]), fmaxf(lg[2], lg[3]));
            #pragma unroll
            for (int s = 1; s < 16; s <<= 1) mx = fmaxf(mx, __shfl_xor(mx, s, 32));
            float sum = 0.0f;
            #pragma unroll
            for (int u = 0; u < 4; ++u) { lg[u] = __expf(lg[u] - mx); sum += lg[u]; }
            #pragma unroll
            for (int s = 1; s < 16; s <<= 1) sum += __shfl_xor(sum, s, 32);
            const float inv = 1.0f / sum;
            #pragma unroll
            for (int u = 0; u < 4; ++u)
                sP[wave][i][u * 16 + l15] = (__bf16)(lg[u] * inv);
        }
    }

    // ---- O = P * V  (K = 64 keys in 2 WMMA steps) ----
    v8f Oa[4][2];
    for (int t = 0; t < 4; ++t)
        for (int u = 0; u < 2; ++u)
            for (int i = 0; i < 8; ++i) Oa[t][u][i] = 0.0f;

    #pragma unroll
    for (int ks = 0; ks < 2; ++ks) {
        FragBF pf[4], vf[2];
        #pragma unroll
        for (int t = 0; t < 4; ++t) {
            const int row = t * 16 + l15;
            pf[t].q[0] = *(const uint4*)&sP[wave][row][ks * 32 + hi * 8];
            pf[t].q[1] = *(const uint4*)&sP[wave][row][ks * 32 + 16 + hi * 8];
        }
        #pragma unroll
        for (int u = 0; u < 2; ++u) {
            const int d = u * 16 + l15;
            vf[u].q[0] = *(const uint4*)&sVT[wave][d][ks * 32 + hi * 8];
            vf[u].q[1] = *(const uint4*)&sVT[wave][d][ks * 32 + 16 + hi * 8];
        }
        #pragma unroll
        for (int t = 0; t < 4; ++t)
            #pragma unroll
            for (int u = 0; u < 2; ++u)
                Oa[t][u] = wmma_bf16(pf[t], vf[u], Oa[t][u]);
    }

    // ---- store O (discard pad rows) ----
    #pragma unroll
    for (int t = 0; t < 4; ++t) {
        #pragma unroll
        for (int u = 0; u < 2; ++u) {
            #pragma unroll
            for (int r = 0; r < 8; ++r) {
                const int row = t * 16 + hi * 8 + r;
                if (row < NTOK) {
                    const int d = u * 16 + l15;
                    o[((size_t)(b * NTOK + row)) * DIM + h * 32 + d] = (__bf16)Oa[t][u][r];
                }
            }
        }
    }
}

// ---------------------------------------------------------------
extern "C" void kernel_launch(void* const* d_in, const int* in_sizes, int n_in,
                              void* d_out, int out_size, void* d_ws, size_t ws_size,
                              hipStream_t stream) {
    const float* x     = (const float*)d_in[0];
    const float* mask  = (const float*)d_in[1];
    // d_in[2] = nw (fixed 64)
    const float* qkv_w = (const float*)d_in[3];
    const float* qkv_b = (const float*)d_in[4];
    const float* rpt   = (const float*)d_in[5];
    const float* out_w = (const float*)d_in[6];
    const float* out_b = (const float*)d_in[7];
    const int*   rpidx = (const int*)d_in[8];

    char* ws = (char*)d_ws;
    const size_t qkvBytes  = (size_t)MTOT * 3072 * 2;   // 308,281,344
    const size_t oBytes    = (size_t)MTOT * DIM  * 2;   // 102,760,448
    const size_t xBytes    = (size_t)MTOT * DIM  * 2;   // 102,760,448
    const size_t qkvwBytes = (size_t)3072 * DIM  * 2;   //   6,291,456

    __bf16* qkv_bf   = (__bf16*)(ws);
    __bf16* o_bf     = (__bf16*)(ws + qkvBytes);
    __bf16* x_bf     = (__bf16*)(ws + qkvBytes + oBytes);
    __bf16* qkvw_bf  = (__bf16*)(ws + qkvBytes + oBytes + xBytes);
    __bf16* outwT_bf = (__bf16*)(ws + qkvBytes + oBytes + xBytes + qkvwBytes);
    float*  biasg    = (float*) (ws + qkvBytes + oBytes + xBytes + qkvwBytes
                                    + (size_t)DIM * DIM * 2);

    // 0. one-shot precision/layout prep (all small vs. 23.3 TB/s)
    {
        int n8 = (MTOT * DIM) / 8;
        k_f32_to_bf16<<<(n8 + 255) / 256, 256, 0, stream>>>(x, x_bf, n8);
    }
    {
        int n8 = (3072 * DIM) / 8;
        k_f32_to_bf16<<<(n8 + 255) / 256, 256, 0, stream>>>(qkv_w, qkvw_bf, n8);
    }
    k_transpose_bf16<<<(DIM * DIM + 255) / 256, 256, 0, stream>>>(out_w, outwT_bf);
    {
        const int tot = NUM_HEADS * NTOK * NTOK;
        k_bias_gather<<<(tot + 255) / 256, 256, 0, stream>>>(rpt, rpidx, biasg);
    }

    // 1. qkv = x @ qkv_w^T + qkv_b   (bf16 WMMA, async double-buffered LDS)
    k_gemm_wmma<true><<<dim3(3072 / 128, MTOT / 128), 256, 0, stream>>>(
        x_bf, qkvw_bf, qkv_b, qkv_bf, MTOT, DIM, 3072);

    // 2. window attention (one wave per (b, h))
    k_attention<<<B_ * (NUM_HEADS / 4), 128, 0, stream>>>(qkv_bf, mask, biasg, o_bf);

    // 3. out = attn_out @ out_w + out_b   (fp32 result straight to d_out)
    k_gemm_wmma<false><<<dim3(DIM / 128, MTOT / 128), 256, 0, stream>>>(
        o_bf, outwT_bf, out_b, (float*)d_out, MTOT, DIM, DIM);
}
